// TextRNN2_19937238188551
// MI455X (gfx1250) — matrix operand order repaired and checked by reading, not compiled, not run on hardware
//
#include <hip/hip_runtime.h>
#include <hip/hip_bf16.h>
#include <math.h>

typedef __attribute__((ext_vector_type(16))) __bf16 v16bf;
typedef __attribute__((ext_vector_type(8)))  __bf16 v8bf;
typedef __attribute__((ext_vector_type(4)))  __bf16 v4bf;
typedef __attribute__((ext_vector_type(8)))  float  v8f;
typedef __attribute__((ext_vector_type(4)))  float  v4f;

#define EMB   512
#define HID   1024
#define BATCH 64
#define SEQ   512
#define RNN_BLOCKS 64          // persistent recurrence blocks, each owns 16 cols of H

static __device__ __forceinline__ v16bf join16(v8bf lo, v8bf hi) {
    v16bf r;
#pragma unroll
    for (int i = 0; i < 8; i++) { r[i] = lo[i]; r[i + 8] = hi[i]; }
    return r;
}

// ---------------- W_aa f32 -> bf16 (once; recurrence then keeps it in registers)
__global__ void convert_waa_kernel(const float* __restrict__ w, v4bf* __restrict__ o) {
    int i = blockIdx.x * blockDim.x + threadIdx.x;      // over HID*HID/4
    v4f v = ((const v4f*)w)[i];
    v4bf r;
#pragma unroll
    for (int j = 0; j < 4; j++) r[j] = (__bf16)v[j];
    o[i] = r;
}

// ---------------- a0 = 0, and reset the device-wide barrier counter (every replay)
__global__ void init_a_kernel(v4bf* __restrict__ a, unsigned* __restrict__ ctr) {
    int i = blockIdx.x * blockDim.x + threadIdx.x;      // over BATCH*HID/4
    v4bf z;
#pragma unroll
    for (int j = 0; j < 4; j++) z[j] = (__bf16)0.0f;
    a[i] = z;
    if (i == 0) *ctr = 0u;
}

// ---------------- Phase 1: xproj[s*64+b][h] = sum_k X[b][s][k]*W_ax[h][k] + b_a[h]
// 128 threads = 4 wave32; 64x64 tile; K-step 32; double-buffered LDS; bf16 WMMA.
__global__ __launch_bounds__(128)
void xproj_gemm_kernel(const float* __restrict__ X, const float* __restrict__ Wax,
                       const float* __restrict__ bias, __bf16* __restrict__ XP) {
    __shared__ __align__(16) __bf16 As[2][64][40];
    __shared__ __align__(16) __bf16 Bs[2][64][40];
    const int t    = threadIdx.x;
    const int lane = t & 31, wave = t >> 5;
    const int r0   = blockIdx.x * 64;            // M block over S*B
    const int n0   = blockIdx.y * 64;            // H block
    const int m    = lane & 15, half = lane >> 4;
    const int KT   = EMB / 32;

    auto stage = [&](int buf, int k0) {
#pragma unroll
        for (int i = 0; i < 4; i++) {
            int chunk = t + i * 128;             // 0..511
            int row = chunk >> 3, c4 = (chunk & 7) * 4;
            int r = r0 + row;
            int s = r >> 6, b = r & 63;          // r = s*64 + b
            const float* xp = &X[((size_t)b * SEQ + s) * EMB + k0 + c4];
            const float* wp = &Wax[(size_t)(n0 + row) * EMB + k0 + c4];
            if (k0 + 64 < EMB) {                 // gfx1250 global_prefetch_b8, 2 tiles ahead
                __builtin_prefetch(xp + 64, 0, 3);
                __builtin_prefetch(wp + 64, 0, 3);
            }
            v4f va = *(const v4f*)xp;
            v4f vb = *(const v4f*)wp;
#pragma unroll
            for (int j = 0; j < 4; j++) {
                As[buf][row][c4 + j] = (__bf16)va[j];
                Bs[buf][row][c4 + j] = (__bf16)vb[j];
            }
        }
    };

    v8f acc[4] = {};
    stage(0, 0);
    __syncthreads();
    for (int kt = 0; kt < KT; kt++) {
        int cur = kt & 1;
        if (kt + 1 < KT) stage(1 - cur, (kt + 1) * 32);
        // A fragment: lane m = M row, lane-half selects K half (ISA 16-bit A layout)
        v8bf alo = *(const v8bf*)&As[cur][wave * 16 + m][half * 8];
        v8bf ahi = *(const v8bf*)&As[cur][wave * 16 + m][16 + half * 8];
        v16bf afrag = join16(alo, ahi);
        // hoist all four B fragments so the 4 WMMAs can issue back-to-back
        v16bf bfrag[4];
#pragma unroll
        for (int nb = 0; nb < 4; nb++) {
            // B fragment: lane m = N col; row-major W == B^T so rows load directly
            v8bf blo = *(const v8bf*)&Bs[cur][nb * 16 + m][half * 8];
            v8bf bhi = *(const v8bf*)&Bs[cur][nb * 16 + m][16 + half * 8];
            bfrag[nb] = join16(blo, bhi);
        }
#pragma unroll
        for (int nb = 0; nb < 4; nb++) {
            acc[nb] = __builtin_amdgcn_wmma_f32_16x16x32_bf16(
                false, afrag, false, bfrag[nb], (short)0, acc[nb], false, false);
        }
        __syncthreads();
    }
#pragma unroll
    for (int nb = 0; nb < 4; nb++) {
#pragma unroll
        for (int r = 0; r < 8; r++) {
            int mg = wave * 16 + r + half * 8;
            int hg = n0 + nb * 16 + m;
            float val = acc[nb][r] + bias[hg];
            XP[(size_t)(r0 + mg) * HID + hg] = (__bf16)val;
        }
    }
}

// ---------------- Phase 2: persistent recurrence. 64 blocks x 128 threads.
// Each wave owns a 16x16 tile (M rows 16w.., block's 16 H cols) and keeps its whole
// K=1024 weight slice as 32 WMMA B-fragments in VGPRs (256 VGPRs) for all 512 steps.
// launch_bounds(128,1): one workgroup per WGP so the compiler can use the full
// CDNA5 VGPR budget and never spill the register-resident weights.
// Per step: 32 x (two 16B A-frag loads from L2-resident hidden state + 1 WMMA),
// tanh epilogue, then a device-wide barrier via a monotonic atomic counter.
__global__ __launch_bounds__(128, 1)
void rnn_persistent_kernel(const __bf16* __restrict__ Wbf,
                           const __bf16* __restrict__ XP,
                           float* __restrict__ out,
                           __bf16* __restrict__ A0,
                           __bf16* __restrict__ A1,
                           unsigned* __restrict__ ctr) {
    const int t    = threadIdx.x;
    const int lane = t & 31, wave = t >> 5;
    const int n0   = blockIdx.x * 16;            // this block's H slice
    const int m    = lane & 15, half = lane >> 4;

    // Hoist the weight slice into registers: B[k][n] = W_aa[n0+n][k]
    v16bf bfr[32];
#pragma unroll
    for (int kt = 0; kt < 32; kt++) {
        const __bf16* wp = &Wbf[(size_t)(n0 + m) * HID + kt * 32 + half * 8];
        bfr[kt] = join16(*(const v8bf*)wp, *(const v8bf*)(wp + 16));
    }

    const int mrow = wave * 16 + m;              // A row this lane fetches
    for (int s = 0; s < SEQ; s++) {
        const __bf16* Acur  = (s & 1) ? A1 : A0;
        __bf16*       Anext = (s & 1) ? A0 : A1;

        v8f acc = {};
#pragma unroll
        for (int kt = 0; kt < 32; kt++) {
            const __bf16* ap = &Acur[(size_t)mrow * HID + kt * 32 + half * 8];
            v16bf afrag = join16(*(const v8bf*)ap, *(const v8bf*)(ap + 16));
            acc = __builtin_amdgcn_wmma_f32_16x16x32_bf16(
                false, afrag, false, bfr[kt], (short)0, acc, false, false);
        }

        const __bf16* XPs  = XP  + (size_t)s * BATCH * HID;
        float*        Outs = out + (size_t)s * BATCH * HID;
#pragma unroll
        for (int r = 0; r < 8; r++) {
            int mg = wave * 16 + r + half * 8;   // batch row
            int hg = n0 + m;                     // H col
            float val = tanhf(acc[r] + (float)XPs[(size_t)mg * HID + hg]);
            Outs[(size_t)mg * HID + hg] = val;
            Anext[(size_t)mg * HID + hg] = (__bf16)val;
            if (s == SEQ - 1)                    // final hidden state
                out[(size_t)SEQ * BATCH * HID + (size_t)mg * HID + hg] = val;
        }

        if (s != SEQ - 1) {                      // device-wide step barrier
            __threadfence();
            __syncthreads();
            if (t == 0) {
                atomicAdd(ctr, 1u);
                unsigned target = (unsigned)RNN_BLOCKS * (unsigned)(s + 1);
                while (atomicAdd(ctr, 0u) < target) __builtin_amdgcn_s_sleep(2);
            }
            __syncthreads();
            __threadfence();
        }
    }
}

extern "C" void kernel_launch(void* const* d_in, const int* in_sizes, int n_in,
                              void* d_out, int out_size, void* d_ws, size_t ws_size,
                              hipStream_t stream) {
    (void)in_sizes; (void)n_in; (void)out_size; (void)ws_size;
    const float* X   = (const float*)d_in[0];   // [B, S, E]
    const float* Wax = (const float*)d_in[1];   // [H, E]
    const float* Waa = (const float*)d_in[2];   // [H, H]
    const float* ba  = (const float*)d_in[3];   // [H]
    float* out = (float*)d_out;                 // [S,B,H] then [B,H]

    char* ws = (char*)d_ws;
    __bf16*   XP  = (__bf16*)ws;                                     // SEQ*BATCH*HID bf16
    __bf16*   Wbf = (__bf16*)(ws + (size_t)SEQ * BATCH * HID * 2);   // HID*HID bf16
    __bf16*   A0  = (__bf16*)((char*)Wbf + (size_t)HID * HID * 2);   // BATCH*HID bf16
    __bf16*   A1  = A0 + (size_t)BATCH * HID;
    unsigned* ctr = (unsigned*)(A1 + (size_t)BATCH * HID);

    convert_waa_kernel<<<(HID * HID / 4) / 256, 256, 0, stream>>>(Waa, (v4bf*)Wbf);
    init_a_kernel<<<(BATCH * HID / 4) / 256, 256, 0, stream>>>((v4bf*)A0, ctr);

    dim3 g1((SEQ * BATCH) / 64, HID / 64);
    xproj_gemm_kernel<<<g1, 128, 0, stream>>>(X, Wax, ba, XP);

    rnn_persistent_kernel<<<RNN_BLOCKS, 128, 0, stream>>>(Wbf, XP, out, A0, A1, ctr);
}
